// DeepCNNLSTM_EncoderMOE_42786464203496
// MI455X (gfx1250) — matrix operand order repaired and checked by reading, not compiled, not run on hardware
//
#include <hip/hip_runtime.h>
#include <hip/hip_bf16.h>

// ---------------------------------------------------------------------------
// DeepCNNLSTM_EncoderMOE for MI455X (gfx1250, wave32, WMMA).
// Heavy ops (all convs, LSTM gate GEMMs) run on v_wmma_f32_16x16x32_f16.
// fp32 storage, f16 GEMM inputs, f32 accumulation.
// Conv weights are pre-packed into per-lane WMMA A-fragment order so the
// conv inner loop is pure {b128 load, ds_load_b128, wmma}.
// ---------------------------------------------------------------------------

typedef __attribute__((ext_vector_type(16))) _Float16 v16h;
typedef __attribute__((ext_vector_type(8)))  _Float16 v8h;
typedef __attribute__((ext_vector_type(8)))  float    v8f;

#define B_    64
#define T_    512
#define CIN_  16
#define E_    8
#define C1_   64
#define C2_   128
#define C3_   256
#define HRT_  128
#define DRT_  64
#define H_    256
#define NC_   10
#define G_    8
#define EPS_  1e-5f
#define TW_   128   // time tile per conv block

// ------------------------------- WMMA helpers ------------------------------
// A fragment: 16x32 f16, row-major source, row stride lda (elements).
// ISA 7.12.2 layout: M = lane&15; lanes 16-31 have K offset +8;
// halves 0..7 -> K = khalf+j, halves 8..15 -> K = 16+khalf+j.
// Requires 16B alignment of (src + m*lda + khalf).
__device__ __forceinline__ v16h load_a_rowmajor(const _Float16* src, int lda) {
  int lane = threadIdx.x & 31;
  const _Float16* p = src + (size_t)(lane & 15) * lda + ((lane >> 4) << 3);
  v8h lo = *(const v8h*)p;
  v8h hi = *(const v8h*)(p + 16);
  v16h a;
#pragma unroll
  for (int j = 0; j < 8; ++j) { a[j] = lo[j]; a[8 + j] = hi[j]; }
  return a;
}

// B fragment: 32x16 (KxN) f16 from an N-major (transposed) source:
// srcT[n*ldb + k]. N=lane&15; lanes 16-31 have K offset +8;
// halves 0..7 -> K = khalf+j, halves 8..15 -> K = 16+khalf+j (16B loads).
__device__ __forceinline__ v16h load_b_nmajor(const _Float16* srcT, int ldb) {
  int lane = threadIdx.x & 31;
  const _Float16* p = srcT + (size_t)(lane & 15) * ldb + ((lane >> 4) << 3);
  v8h lo = *(const v8h*)p;
  v8h hi = *(const v8h*)(p + 16);
  v16h b;
#pragma unroll
  for (int j = 0; j < 8; ++j) { b[j] = lo[j]; b[8 + j] = hi[j]; }
  return b;
}

__device__ __forceinline__ v8f wmma_f16(v16h a, v16h b, v8f c) {
  return __builtin_amdgcn_wmma_f32_16x16x32_f16(false, a, false, b, (short)0, c,
                                                false, false);
}

// ------------------------ conv weight pre-packing --------------------------
// Pack w (Cout, Cin, 5) f32 into A-fragment order f16:
//   wp[((tile*nK + kc)*5 + tap)*512 + lane*16 + j]
// holding exactly the 16 halves lane `lane` needs for the 16x32 A tile of
// (co-tile `tile`, ci-chunk `kc`, tap). Zero-pads k >= Cin.
__global__ void pack_convw_k(const float* __restrict__ w,
                             _Float16* __restrict__ wp, int Cin, int Cout,
                             int nK) {
  int total = (Cout >> 4) * nK * 5 * 512;
  for (int idx = blockIdx.x * 256 + threadIdx.x; idx < total;
       idx += gridDim.x * 256) {
    int j = idx & 15;
    int lane = (idx >> 4) & 31;
    int r = idx >> 9;
    int tap = r % 5; r /= 5;
    int kc = r % nK;
    int tile = r / nK;
    int m = lane & 15;
    int khalf = (lane >> 4) << 3;
    int k = kc * 32 + ((j < 8) ? (khalf + j) : (8 + khalf + j));
    float v = 0.f;
    if (k < Cin) v = w[((size_t)(tile * 16 + m) * Cin + k) * 5 + tap];
    wp[idx] = (_Float16)v;
  }
}

// ------------------------- conv1d (implicit GEMM) --------------------------
// grid: (T/128, Cout/16, B); block 256 (8 waves, each a 16co x 16t tile).
__global__ __launch_bounds__(256) void conv1d_wmma_k(
    const float* __restrict__ in, const _Float16* __restrict__ wp,
    float* __restrict__ out, int Cin, int Cout, int nK) {
  __shared__ _Float16 xs[(TW_ + 4) * 128];  // t-major, channel-contiguous rows
  const int b   = blockIdx.z;
  const int co0 = blockIdx.y << 4;
  const int t0  = blockIdx.x * TW_;
  const int CinP = nK << 5;
  const int nst = CinP * (TW_ + 4);
  // stage input tile (+halo) as f16; coalesced in t, zero pad ci>=Cin / OOB t
  for (int i = threadIdx.x; i < nst; i += 256) {
    int ci = i / (TW_ + 4), tl = i - ci * (TW_ + 4);
    int t = t0 + tl - 2;
    float v = 0.f;
    if (ci < Cin && t >= 0 && t < T_) v = in[((size_t)b * Cin + ci) * T_ + t];
    xs[tl * CinP + ci] = (_Float16)v;
  }
  __syncthreads();
  const int wv = threadIdx.x >> 5;
  const int tsub = wv << 4;
  const int lane = threadIdx.x & 31;
  const _Float16* wtile = wp + (size_t)blockIdx.y * nK * 5 * 512 + lane * 16;
  v8f acc = {};
  for (int kc = 0; kc < nK; ++kc) {
#pragma unroll
    for (int tap = 0; tap < 5; ++tap) {
      v16h a  = *(const v16h*)(wtile + (size_t)(kc * 5 + tap) * 512);
      v16h bm = load_b_nmajor(xs + (tsub + tap) * CinP + (kc << 5), CinP);
      acc = wmma_f16(a, bm, acc);
    }
  }
  const int n = lane & 15;
  const int mofs = (lane >> 4) << 3;
#pragma unroll
  for (int r = 0; r < 8; ++r) {
    int m = r + mofs;
    out[((size_t)b * Cout + co0 + m) * T_ + t0 + tsub + n] = acc[r];
  }
}

// ------------------------------ GroupNorm ----------------------------------
__global__ __launch_bounds__(256) void gn_stats_k(const float* __restrict__ x,
                                                  float* __restrict__ stats,
                                                  int C) {
  const int bg = blockIdx.x;               // B*G blocks
  const int b = bg / G_, g = bg % G_;
  const int Cg = C / G_;
  const float* base = x + ((size_t)b * C + g * Cg) * T_;
  const int n = Cg * T_;
  float s = 0.f, q = 0.f;
  for (int i = threadIdx.x; i < n; i += 256) { float v = base[i]; s += v; q += v * v; }
  __shared__ float shs[256], shq[256];
  shs[threadIdx.x] = s; shq[threadIdx.x] = q;
  __syncthreads();
  for (int k = 128; k > 0; k >>= 1) {
    if (threadIdx.x < k) { shs[threadIdx.x] += shs[threadIdx.x + k];
                           shq[threadIdx.x] += shq[threadIdx.x + k]; }
    __syncthreads();
  }
  if (threadIdx.x == 0) {
    float m = shs[0] / n;
    float var = shq[0] / n - m * m;
    stats[bg * 2]     = m;
    stats[bg * 2 + 1] = rsqrtf(var + EPS_);
  }
}

__global__ void gn_gelu_k(float* __restrict__ x, const float* __restrict__ gamma,
                          const float* __restrict__ beta,
                          const float* __restrict__ stats, int C) {
  const size_t total = (size_t)B_ * C * T_;
  const int Cg = C / G_;
  for (size_t i = blockIdx.x * 256ull + threadIdx.x; i < total;
       i += (size_t)gridDim.x * 256ull) {
    int c = (int)((i / T_) % C);
    int b = (int)(i / ((size_t)C * T_));
    int g = c / Cg;
    float m  = stats[(b * G_ + g) * 2];
    float rs = stats[(b * G_ + g) * 2 + 1];
    float xn = (x[i] - m) * rs * gamma[c] + beta[c];
    x[i] = 0.5f * xn * (1.f + erff(xn * 0.70710678118f));  // exact GELU
  }
}

// ---------------------------- router pieces --------------------------------
__global__ void gap_k(const float* __restrict__ x, float* __restrict__ rh) {
  int idx = blockIdx.x * 256 + threadIdx.x;       // B*HRT threads
  if (idx >= B_ * HRT_) return;
  const float* p = x + (size_t)idx * T_;
  float s = 0.f;
  for (int t = 0; t < T_; ++t) s += p[t];
  rh[idx] = s * (1.f / T_);
}

__global__ __launch_bounds__(256) void router_mlp_k(
    const float* __restrict__ rh, const float* __restrict__ m1W,
    const float* __restrict__ m1b, const float* __restrict__ lng,
    const float* __restrict__ lnb, const float* __restrict__ m2W,
    const float* __restrict__ m2b, const float* __restrict__ gW,
    const float* __restrict__ gb, float* __restrict__ wh) {
  __shared__ float z1[B_ * HRT_];
  __shared__ float rr[B_ * DRT_];
  const int tid = threadIdx.x;
  for (int i = tid; i < B_ * HRT_; i += 256) {
    int b = i >> 7, j = i & 127;
    float s = m1b[j];
    const float* rbv = rh + b * HRT_;
    for (int k = 0; k < HRT_; ++k) s += rbv[k] * m1W[k * HRT_ + j];
    z1[i] = s;
  }
  __syncthreads();
  if (tid < B_) {                         // LayerNorm + GELU per row
    float* z = z1 + tid * HRT_;
    float m = 0.f;
    for (int k = 0; k < HRT_; ++k) m += z[k];
    m *= (1.f / HRT_);
    float v = 0.f;
    for (int k = 0; k < HRT_; ++k) { float d = z[k] - m; v += d * d; }
    v *= (1.f / HRT_);
    float rs = rsqrtf(v + EPS_);
    for (int k = 0; k < HRT_; ++k) {
      float xn = (z[k] - m) * rs * lng[k] + lnb[k];
      z[k] = 0.5f * xn * (1.f + erff(xn * 0.70710678118f));
    }
  }
  __syncthreads();
  for (int i = tid; i < B_ * DRT_; i += 256) {
    int b = i >> 6, dd = i & 63;
    float s = m2b[dd];
    const float* z = z1 + b * HRT_;
    for (int k = 0; k < HRT_; ++k) s += z[k] * m2W[k * DRT_ + dd];
    rr[i] = s;
  }
  __syncthreads();
  if (tid < B_) {                         // gate: softmax -> top2 -> renorm
    float lg[E_];
    float mx = -1e30f;
    for (int e = 0; e < E_; ++e) {
      float s = gb[e];
      const float* r = rr + tid * DRT_;
      for (int k = 0; k < DRT_; ++k) s += r[k] * gW[k * E_ + e];
      lg[e] = s;
      mx = fmaxf(mx, s);
    }
    float sum = 0.f;
    for (int e = 0; e < E_; ++e) { lg[e] = __expf(lg[e] - mx); sum += lg[e]; }
    float inv = 1.f / sum;
    for (int e = 0; e < E_; ++e) lg[e] *= inv;
    int i1 = 0;
    for (int e = 1; e < E_; ++e) if (lg[e] > lg[i1]) i1 = e;
    int i2 = (i1 == 0) ? 1 : 0;
    for (int e = 0; e < E_; ++e) if (e != i1 && lg[e] > lg[i2]) i2 = e;
    float denom = lg[i1] + lg[i2] + 1e-9f;
    for (int e = 0; e < E_; ++e)
      wh[tid * E_ + e] = (e == i1 || e == i2) ? lg[e] / denom : 0.f;
  }
}

// ------------------------------ mixture ------------------------------------
__global__ void mix_axpy_k(float* __restrict__ acc, const float* __restrict__ ex,
                           const float* __restrict__ wh, int e) {
  const size_t total = (size_t)B_ * C3_ * T_;
  for (size_t i = blockIdx.x * 256ull + threadIdx.x; i < total;
       i += (size_t)gridDim.x * 256ull) {
    int b = (int)(i / ((size_t)C3_ * T_));
    acc[i] += wh[b * E_ + e] * ex[i];
  }
}

// ------------------------------ conversions --------------------------------
// src RxC row-major -> dst CxR row-major (N-major for B fragments), f16.
__global__ void transpose_cvt_k(const float* __restrict__ s,
                                _Float16* __restrict__ d, int R, int C) {
  int total = R * C;
  for (int i = blockIdx.x * 256 + threadIdx.x; i < total; i += gridDim.x * 256) {
    int r = i / C, c = i - r * C;
    d[(size_t)c * R + r] = (_Float16)s[i];
  }
}

// acc (B,C3,T) f32 -> xT (B,T,C3) f16 (A-matrix friendly for LSTM x-GEMM)
__global__ void cvt_hmix_k(const float* __restrict__ acc,
                           _Float16* __restrict__ xT) {
  const size_t total = (size_t)B_ * C3_ * T_;
  for (size_t i = blockIdx.x * 256ull + threadIdx.x; i < total;
       i += (size_t)gridDim.x * 256ull) {
    size_t b = i / ((size_t)C3_ * T_);
    size_t rem = i - b * (size_t)C3_ * T_;
    size_t c = rem / T_, t = rem - c * T_;
    xT[(b * T_ + t) * C3_ + c] = (_Float16)acc[i];
  }
}

// ------------------------------- LSTM --------------------------------------
// One persistent workgroup: 1024 threads = 32 waves. Each wave owns two
// 16(batch)x16(hidden) quads; for each quad it computes all four gate tiles
// (i,f,g,o) so the cell update is wave-local. c stays in VGPRs, h in LDS f16.
__global__ __launch_bounds__(1024) void lstm_k(
    const _Float16* __restrict__ xT,    // [B][T][H_] f16
    const _Float16* __restrict__ WihT,  // [4H][H] f16 (N-major)
    const _Float16* __restrict__ WhhT,  // [4H][H] f16 (N-major)
    const float* __restrict__ bias,     // [4H]
    float* __restrict__ hT) {           // [B][H] f32 (final hidden)
  __shared__ _Float16 hls[B_ * H_];     // 32KB
  const int tid = threadIdx.x;
  const int wv = tid >> 5;
  const int lane = tid & 31;
  const int q0 = wv << 1;               // quads 2*wv, 2*wv+1 (same m-tile)
  const int m0 = (q0 >> 4) << 4;        // batch tile base
  const int n = lane & 15;
  const int mofs = (lane >> 4) << 3;

  for (int i = tid; i < B_ * H_; i += 1024) hls[i] = (_Float16)0.f;

  v8f creg[2];
#pragma unroll
  for (int qi = 0; qi < 2; ++qi)
#pragma unroll
    for (int r = 0; r < 8; ++r) creg[qi][r] = 0.f;

  float bquad[2][4];
#pragma unroll
  for (int qi = 0; qi < 2; ++qi) {
    int j0 = ((q0 + qi) & 15) << 4;
#pragma unroll
    for (int g = 0; g < 4; ++g) bquad[qi][g] = bias[g * H_ + j0 + n];
  }
  __syncthreads();

  for (int t = 0; t < T_; ++t) {
    v8f accq[2][4];
#pragma unroll
    for (int qi = 0; qi < 2; ++qi)
#pragma unroll
      for (int g = 0; g < 4; ++g)
#pragma unroll
        for (int r = 0; r < 8; ++r) accq[qi][g][r] = bquad[qi][g];

    const _Float16* xrow = xT + (size_t)m0 * T_ * H_ + (size_t)t * H_;
    for (int k0 = 0; k0 < H_; k0 += 32) {
      v16h ax = load_a_rowmajor(xrow + k0, T_ * H_);
      v16h ah = load_a_rowmajor(hls + m0 * H_ + k0, H_);
#pragma unroll
      for (int qi = 0; qi < 2; ++qi) {
        int j0 = ((q0 + qi) & 15) << 4;
#pragma unroll
        for (int g = 0; g < 4; ++g) {
          v16h bx = load_b_nmajor(WihT + (size_t)(g * H_ + j0) * H_ + k0, H_);
          accq[qi][g] = wmma_f16(ax, bx, accq[qi][g]);
          v16h bh = load_b_nmajor(WhhT + (size_t)(g * H_ + j0) * H_ + k0, H_);
          accq[qi][g] = wmma_f16(ah, bh, accq[qi][g]);
        }
      }
    }
    __syncthreads();  // all reads of hls done before overwrite
#pragma unroll
    for (int qi = 0; qi < 2; ++qi) {
      int j0 = ((q0 + qi) & 15) << 4;
#pragma unroll
      for (int r = 0; r < 8; ++r) {
        float iv = 1.f / (1.f + __expf(-accq[qi][0][r]));
        float fv = 1.f / (1.f + __expf(-accq[qi][1][r]));
        float gv = tanhf(accq[qi][2][r]);
        float ov = 1.f / (1.f + __expf(-accq[qi][3][r]));
        float cv = fv * creg[qi][r] + iv * gv;
        creg[qi][r] = cv;
        float hv = ov * tanhf(cv);
        int row = m0 + r + mofs;
        hls[row * H_ + j0 + n] = (_Float16)hv;
        if (t == T_ - 1) hT[row * H_ + j0 + n] = hv;
      }
    }
    __syncthreads();
  }
}

// ----------------------------- classifier ----------------------------------
__global__ void classifier_k(const float* __restrict__ hT,
                             const float* __restrict__ Wc,
                             const float* __restrict__ bc,
                             float* __restrict__ out) {
  int idx = blockIdx.x * 256 + threadIdx.x;
  if (idx >= B_ * NC_) return;
  int b = idx / NC_, n = idx - b * NC_;
  float s = bc[n];
  const float* h = hT + b * H_;
  for (int k = 0; k < H_; ++k) s += h[k] * Wc[k * NC_ + n];
  out[idx] = s;
}

// ------------------------------ host side ----------------------------------
static void conv_gn_gelu_stage(const float* in, const _Float16* wp, float* out,
                               int Cin, int Cout, int nK, const float* gamma,
                               const float* beta, float* stats,
                               hipStream_t s) {
  dim3 g(T_ / TW_, Cout / 16, B_);
  conv1d_wmma_k<<<g, 256, 0, s>>>(in, wp, out, Cin, Cout, nK);
  gn_stats_k<<<B_ * G_, 256, 0, s>>>(out, stats, Cout);
  gn_gelu_k<<<2048, 256, 0, s>>>(out, gamma, beta, stats, Cout);
}

extern "C" void kernel_launch(void* const* d_in, const int* in_sizes, int n_in,
                              void* d_out, int out_size, void* d_ws,
                              size_t ws_size, hipStream_t stream) {
  (void)in_sizes; (void)n_in; (void)out_size; (void)ws_size;

  const float* x    = (const float*)d_in[0];
  const float* eW1  = (const float*)d_in[1];
  const float* eg1  = (const float*)d_in[2];
  const float* eb1  = (const float*)d_in[3];
  const float* eW2  = (const float*)d_in[4];
  const float* eg2  = (const float*)d_in[5];
  const float* eb2  = (const float*)d_in[6];
  const float* eW3  = (const float*)d_in[7];
  const float* eg3  = (const float*)d_in[8];
  const float* eb3  = (const float*)d_in[9];
  const float* sW1  = (const float*)d_in[10];
  const float* sg1  = (const float*)d_in[11];
  const float* sb1  = (const float*)d_in[12];
  const float* sW2  = (const float*)d_in[13];
  const float* sg2  = (const float*)d_in[14];
  const float* sb2  = (const float*)d_in[15];
  const float* sW3  = (const float*)d_in[16];
  const float* sg3  = (const float*)d_in[17];
  const float* sb3  = (const float*)d_in[18];
  const float* rW   = (const float*)d_in[19];
  const float* rg   = (const float*)d_in[20];
  const float* rb   = (const float*)d_in[21];
  const float* rm1W = (const float*)d_in[22];
  const float* rm1b = (const float*)d_in[23];
  const float* rlng = (const float*)d_in[24];
  const float* rlnb = (const float*)d_in[25];
  const float* rm2W = (const float*)d_in[26];
  const float* rm2b = (const float*)d_in[27];
  const float* gW   = (const float*)d_in[28];
  const float* gb   = (const float*)d_in[29];
  const float* W_ih = (const float*)d_in[30];
  const float* W_hh = (const float*)d_in[31];
  const float* b_l  = (const float*)d_in[32];
  const float* Wc   = (const float*)d_in[33];
  const float* bc   = (const float*)d_in[34];

  // packed fragment sizes (halves) per conv layer
  const int nK1 = 1, nK2 = 2, nK3 = 4, nKr = 1;
  const size_t pk1 = (size_t)(C1_ / 16) * nK1 * 5 * 512;   // 10240
  const size_t pk2 = (size_t)(C2_ / 16) * nK2 * 5 * 512;   // 40960
  const size_t pk3 = (size_t)(C3_ / 16) * nK3 * 5 * 512;   // 163840
  const size_t pkr = (size_t)(HRT_ / 16) * nKr * 5 * 512;  // 20480

  // bump allocator over workspace (~115 MB total)
  char* base = (char*)d_ws;
  size_t off = 0;
  auto take = [&](size_t bytes) -> void* {
    off = (off + 255) & ~(size_t)255;
    void* p = base + off;
    off += bytes;
    return p;
  };
  _Float16* eWp1 = (_Float16*)take(2ull * E_ * pk1);
  _Float16* eWp2 = (_Float16*)take(2ull * E_ * pk2);
  _Float16* eWp3 = (_Float16*)take(2ull * E_ * pk3);
  _Float16* sWp1 = (_Float16*)take(2ull * pk1);
  _Float16* sWp2 = (_Float16*)take(2ull * pk2);
  _Float16* sWp3 = (_Float16*)take(2ull * pk3);
  _Float16* rWp  = (_Float16*)take(2ull * pkr);
  _Float16* WihT = (_Float16*)take(2ull * 4 * H_ * H_);
  _Float16* WhhT = (_Float16*)take(2ull * 4 * H_ * H_);
  float* bufA  = (float*)take(4ull * B_ * C1_ * T_);
  float* bufB  = (float*)take(4ull * B_ * C2_ * T_);
  float* bufC  = (float*)take(4ull * B_ * C3_ * T_);
  float* accF  = (float*)take(4ull * B_ * C3_ * T_);
  _Float16* accH = (_Float16*)take(2ull * B_ * T_ * C3_);
  float* stats = (float*)take(4ull * B_ * G_ * 2);
  float* rh    = (float*)take(4ull * B_ * HRT_);
  float* whard = (float*)take(4ull * B_ * E_);
  float* hTbuf = (float*)take(4ull * B_ * H_);

  // --- 1) weight prep: pack conv weights as WMMA A-fragments (f16) ---
  for (int e = 0; e < E_; ++e) {
    pack_convw_k<<<64, 256, 0, stream>>>(eW1 + (size_t)e * C1_ * CIN_ * 5,
                                         eWp1 + (size_t)e * pk1, CIN_, C1_, nK1);
    pack_convw_k<<<160, 256, 0, stream>>>(eW2 + (size_t)e * C2_ * C1_ * 5,
                                          eWp2 + (size_t)e * pk2, C1_, C2_, nK2);
    pack_convw_k<<<640, 256, 0, stream>>>(eW3 + (size_t)e * C3_ * C2_ * 5,
                                          eWp3 + (size_t)e * pk3, C2_, C3_, nK3);
  }
  pack_convw_k<<<64, 256, 0, stream>>>(sW1, sWp1, CIN_, C1_, nK1);
  pack_convw_k<<<160, 256, 0, stream>>>(sW2, sWp2, C1_, C2_, nK2);
  pack_convw_k<<<640, 256, 0, stream>>>(sW3, sWp3, C2_, C3_, nK3);
  pack_convw_k<<<80, 256, 0, stream>>>(rW, rWp, CIN_, HRT_, nKr);
  transpose_cvt_k<<<1024, 256, 0, stream>>>(W_ih, WihT, H_, 4 * H_);
  transpose_cvt_k<<<1024, 256, 0, stream>>>(W_hh, WhhT, H_, 4 * H_);

  // --- 2) router (conv -> GN -> GELU -> GAP -> MLP -> top-2 gate) ---
  conv_gn_gelu_stage(x, rWp, bufB, CIN_, HRT_, nKr, rg, rb, stats, stream);
  gap_k<<<(B_ * HRT_ + 255) / 256, 256, 0, stream>>>(bufB, rh);
  router_mlp_k<<<1, 256, 0, stream>>>(rh, rm1W, rm1b, rlng, rlnb, rm2W, rm2b,
                                      gW, gb, whard);

  // --- 3) shared expert -> accF ---
  conv_gn_gelu_stage(x, sWp1, bufA, CIN_, C1_, nK1, sg1, sb1, stats, stream);
  conv_gn_gelu_stage(bufA, sWp2, bufB, C1_, C2_, nK2, sg2, sb2, stats, stream);
  conv_gn_gelu_stage(bufB, sWp3, accF, C2_, C3_, nK3, sg3, sb3, stats, stream);

  // --- 4) experts, accumulated with gate weights (buffers reused) ---
  for (int e = 0; e < E_; ++e) {
    conv_gn_gelu_stage(x, eWp1 + (size_t)e * pk1, bufA, CIN_, C1_, nK1,
                       eg1 + e * C1_, eb1 + e * C1_, stats, stream);
    conv_gn_gelu_stage(bufA, eWp2 + (size_t)e * pk2, bufB, C1_, C2_, nK2,
                       eg2 + e * C2_, eb2 + e * C2_, stats, stream);
    conv_gn_gelu_stage(bufB, eWp3 + (size_t)e * pk3, bufC, C2_, C3_, nK3,
                       eg3 + e * C3_, eb3 + e * C3_, stats, stream);
    mix_axpy_k<<<2048, 256, 0, stream>>>(accF, bufC, whard, e);
  }

  // --- 5) LSTM over time (persistent single-WGP kernel) ---
  cvt_hmix_k<<<2048, 256, 0, stream>>>(accF, accH);
  lstm_k<<<1, 1024, 0, stream>>>(accH, WihT, WhhT, b_l, hTbuf);

  // --- 6) classifier head ---
  classifier_k<<<(B_ * NC_ + 255) / 256, 256, 0, stream>>>(hTbuf, Wc, bc,
                                                           (float*)d_out);
}
